// TwoStepSoftmax_40235253629259
// MI455X (gfx1250) — compile-verified, matrix operand order checked by reading, not run on hardware
//
#include <hip/hip_runtime.h>
#include <hip/hip_bf16.h>
#include <math.h>

#define VOCAB 50257
#define NTOK  2048
#define DIM   1024

#define BM 64
#define BN 128
#define BK 32
#define LDT 40   // padded LDS row stride (elements) -> conflict-free b128 reads

typedef __attribute__((ext_vector_type(16))) __bf16 v16bf;
typedef __attribute__((ext_vector_type(8)))  __bf16 v8bf;
typedef __attribute__((ext_vector_type(4)))  __bf16 v4bf;
typedef __attribute__((ext_vector_type(8)))  float  v8f;

// ---------------------------------------------------------------------------
// GEMM: out[row, colOffset + col] = sum_k X[row,k] * W[col,k]   (bf16 WMMA)
// Block tile 64x128, 8 waves, each wave computes a 16x64 sub-tile (4 WMMAs
// per K-step sharing one A fragment). grid = (rowTiles, vocabTiles) with row
// tiles fastest so each W tile is reused by all 32 row blocks via L2 and the
// 206 MB of weights stream from HBM exactly once.
// ---------------------------------------------------------------------------
__global__ __launch_bounds__(256)
void asoft_gemm_bf16(const float* __restrict__ X, const float* __restrict__ W,
                     float* __restrict__ out, int Vi, int colOffset) {
  __shared__ __bf16 lsA[BM * LDT];
  __shared__ __bf16 lsB[BN * LDT];

  const int tid  = threadIdx.x;
  const int lane = tid & 31;
  const int wave = tid >> 5;      // 0..7
  const int wm   = wave & 3;      // M sub-tile (16 rows)
  const int wn   = wave >> 2;     // N sub-tile (64 cols)
  const int half = lane >> 4;
  const int l16  = lane & 15;

  const int rowBase = blockIdx.x * BM;
  const int colBase = blockIdx.y * BN;

  v8f acc[4];
#pragma unroll
  for (int i = 0; i < 4; ++i) acc[i] = (v8f){};

  for (int k0 = 0; k0 < DIM; k0 += BK) {
    // ---- cooperative load: A (64x32) and B (128x32), f32 -> bf16 into LDS
#pragma unroll
    for (int j = 0; j < 2; ++j) {          // A: 512 float4
      const int idx = tid + 256 * j;
      const int r   = idx >> 3;
      const int c   = (idx & 7) * 4;
      const float4 av = *(const float4*)(X + (size_t)(rowBase + r) * DIM + k0 + c);
      v4bf ap;
      ap[0] = (__bf16)av.x; ap[1] = (__bf16)av.y;
      ap[2] = (__bf16)av.z; ap[3] = (__bf16)av.w;
      *(v4bf*)(&lsA[r * LDT + c]) = ap;
    }
#pragma unroll
    for (int j = 0; j < 4; ++j) {          // B: 1024 float4
      const int idx = tid + 256 * j;
      const int r   = idx >> 3;
      const int c   = (idx & 7) * 4;
      int n = colBase + r;                 // clamp ragged last cluster
      if (n > Vi - 1) n = Vi - 1;
      const float4 bv = *(const float4*)(W + (size_t)n * DIM + k0 + c);
      v4bf bp;
      bp[0] = (__bf16)bv.x; bp[1] = (__bf16)bv.y;
      bp[2] = (__bf16)bv.z; bp[3] = (__bf16)bv.w;
      *(v4bf*)(&lsB[r * LDT + c]) = bp;
    }
    __syncthreads();

    // ---- A fragment (ISA 16-bit A layout: lanes 0-15 hold K 0-7 | 16-23,
    //      lanes 16-31 hold K 8-15 | 24-31; two contiguous b128 LDS reads)
    v16bf afrag;
    {
      const __bf16* p = &lsA[(wm * 16 + l16) * LDT + 8 * half];
      ((v8bf*)&afrag)[0] = *(const v8bf*)p;
      ((v8bf*)&afrag)[1] = *(const v8bf*)(p + 16);
    }
    // ---- 4 B fragments + 4 WMMAs
#pragma unroll
    for (int bsub = 0; bsub < 4; ++bsub) {
      v16bf bfrag;
      const __bf16* p = &lsB[(wn * 64 + bsub * 16 + l16) * LDT + 8 * half];
      ((v8bf*)&bfrag)[0] = *(const v8bf*)p;
      ((v8bf*)&bfrag)[1] = *(const v8bf*)(p + 16);
      acc[bsub] = __builtin_amdgcn_wmma_f32_16x16x32_bf16(
          false, afrag, false, bfrag, (short)0, acc[bsub], false, false);
    }
    __syncthreads();
  }

  // ---- store logits (non-temporal: keep L2 for weight-tile reuse)
  const int gr0 = rowBase + wm * 16 + half * 8;
#pragma unroll
  for (int bsub = 0; bsub < 4; ++bsub) {
    const int gcol = colBase + wn * 64 + bsub * 16 + l16;
    if (gcol < Vi) {
      float* p = out + (size_t)gr0 * VOCAB + colOffset + gcol;
#pragma unroll
      for (int r = 0; r < 8; ++r)
        __builtin_nontemporal_store(acc[bsub][r], p + (size_t)r * VOCAB);
    }
  }
}

// ---------------------------------------------------------------------------
// Head: 3-way projection + fp32 log_softmax -> head_lp[NTOK][3] in workspace
// ---------------------------------------------------------------------------
__global__ __launch_bounds__(128)
void head_logprob(const float* __restrict__ X, const float* __restrict__ HW,
                  float* __restrict__ head_lp) {
  __shared__ float sh[3 * 128];
  const int row = blockIdx.x;
  const int tid = threadIdx.x;
  float d0 = 0.f, d1 = 0.f, d2 = 0.f;
  const float* xr = X + (size_t)row * DIM;
  for (int k = tid; k < DIM; k += 128) {
    const float xv = xr[k];
    d0 += xv * HW[k];
    d1 += xv * HW[DIM + k];
    d2 += xv * HW[2 * DIM + k];
  }
  sh[tid] = d0; sh[128 + tid] = d1; sh[256 + tid] = d2;
  __syncthreads();
  for (int s = 64; s > 0; s >>= 1) {
    if (tid < s) {
      sh[tid]       += sh[tid + s];
      sh[128 + tid] += sh[128 + tid + s];
      sh[256 + tid] += sh[256 + tid + s];
    }
    __syncthreads();
  }
  if (tid == 0) {
    const float l0 = sh[0], l1 = sh[128], l2 = sh[256];
    const float m   = fmaxf(l0, fmaxf(l1, l2));
    const float lse = m + logf(expf(l0 - m) + expf(l1 - m) + expf(l2 - m));
    head_lp[row * 3 + 0] = l0 - lse;
    head_lp[row * 3 + 1] = l1 - lse;
    head_lp[row * 3 + 2] = l2 - lse;
  }
}

// ---------------------------------------------------------------------------
// Per (row, cluster): one-pass online (max, sumexp) reduction, then rewrite
// the row segment in place:  out = logit - max - log(sum) + head_lp.
// ---------------------------------------------------------------------------
__global__ __launch_bounds__(256)
void logsoftmax_fix(float* __restrict__ out, const float* __restrict__ head_lp,
                    int Vi, int colOffset, int cluster) {
  __shared__ float redM[8];
  __shared__ float redS[8];
  const int row  = blockIdx.x;
  const int tid  = threadIdx.x;
  const int lane = tid & 31;
  const int wave = tid >> 5;
  float* base = out + (size_t)row * VOCAB + colOffset;

  // single sweep: online max + rescaled sum of exp
  float m = -INFINITY;
  float s = 0.f;
  for (int i = tid; i < Vi; i += 256) {
    const float v = base[i];
    if (v > m) {
      s = s * expf(m - v) + 1.f;
      m = v;
    } else {
      s += expf(v - m);
    }
  }
  // lane merge within wave32
  for (int t = 16; t > 0; t >>= 1) {
    const float om = __shfl_xor(m, t, 32);
    const float os = __shfl_xor(s, t, 32);
    const float nm = fmaxf(m, om);
    s = s * expf(m - nm) + os * expf(om - nm);
    m = nm;
  }
  if (lane == 0) { redM[wave] = m; redS[wave] = s; }
  __syncthreads();
  if (tid == 0) {
    float mm = redM[0], ss = redS[0];
    for (int w = 1; w < 8; ++w) {
      const float nm = fmaxf(mm, redM[w]);
      ss = ss * expf(mm - nm) + redS[w] * expf(redM[w] - nm);
      mm = nm;
    }
    redM[0] = head_lp[row * 3 + cluster] - mm - logf(ss);
  }
  __syncthreads();
  const float shift = redM[0];

  // rewrite in place
  for (int i = tid; i < Vi; i += 256) base[i] += shift;
}

// ---------------------------------------------------------------------------
extern "C" void kernel_launch(void* const* d_in, const int* in_sizes, int n_in,
                              void* d_out, int out_size, void* d_ws, size_t ws_size,
                              hipStream_t stream) {
  (void)in_sizes; (void)n_in; (void)out_size; (void)ws_size;
  const float* x  = (const float*)d_in[0];
  const float* hw = (const float*)d_in[1];
  const float* w0 = (const float*)d_in[2];
  const float* w1 = (const float*)d_in[3];
  const float* w2 = (const float*)d_in[4];
  float* out      = (float*)d_out;
  float* head_lp  = (float*)d_ws;   // NTOK*3 floats = 24 KB

  head_logprob<<<NTOK, 128, 0, stream>>>(x, hw, head_lp);

  const int    Vs[3] = {20000, 20000, 10257};
  const float* Ws[3] = {w0, w1, w2};
  int off = 0;
  for (int c = 0; c < 3; ++c) {
    dim3 grid(NTOK / BM, (Vs[c] + BN - 1) / BN);
    asoft_gemm_bf16<<<grid, 256, 0, stream>>>(x, Ws[c], out, Vs[c], off);
    off += Vs[c];
  }
  off = 0;
  for (int c = 0; c < 3; ++c) {
    logsoftmax_fix<<<NTOK, 256, 0, stream>>>(out, head_lp, Vs[c], off, c);
    off += Vs[c];
  }
}
